// TransformerEncoderLayer_25563645345944
// MI455X (gfx1250) — matrix operand order repaired and checked by reading, not compiled, not run on hardware
//
#include <hip/hip_runtime.h>
#include <hip/hip_bf16.h>
#include <math.h>

// ---------------------------------------------------------------------------
// Transformer encoder (5 layers, N=4096, d_model=128, 8 heads x 128 dim)
// bf16 WMMA (v_wmma_f32_16x16x32_bf16) everywhere; flash attention.
// All working data (~40MB) fits in MI455X's 192MB L2 -> fragments loaded
// straight from global; LDS only used for the P C-frag -> A-frag transpose.
// GEMM: 32x32 tile per wave (2x2 wmma accumulators) -> 1 wmma : 2 b128 loads.
// ---------------------------------------------------------------------------

typedef __attribute__((ext_vector_type(16))) __bf16 v16bf;
typedef __attribute__((ext_vector_type(8)))  float  v8f;
typedef unsigned short u16;
typedef unsigned int   u32;

#define NTOK   4096
#define DM     128
#define HEADS  8
#define NQK    1024          // DM per head * HEADS columns of wq/wk/wv

__device__ __forceinline__ u16 f2bf(float f) {
    u32 u = __float_as_uint(f);
    u32 r = u + 0x7FFFu + ((u >> 16) & 1u);   // round-to-nearest-even
    return (u16)(r >> 16);
}

union FragAB { v16bf v; uint4 q[2]; u16 s[16]; };
union FragC  { v8f   v; float f[8]; };

__device__ __forceinline__ v8f wmma_bf16(v16bf a, v16bf b, v8f c) {
    return __builtin_amdgcn_wmma_f32_16x16x32_bf16(
        /*neg_a=*/false, a, /*neg_b=*/false, b,
        /*c_mod=*/(short)0, c, /*reuse_a=*/false, /*reuse_b=*/false);
}

// A fragment (16x32 bf16, MxK): lane r (=lane&15) holds row m0+r.
// elements e<8 -> k = k0 + 8*half + e ; e>=8 -> k = k0 + 16 + 8*half + (e-8)
__device__ __forceinline__ v16bf load_a_frag(const u16* __restrict__ A, int lda,
                                             int m0, int k0, int lane) {
    int r = lane & 15, half = lane >> 4;
    const u16* p = A + (size_t)(m0 + r) * lda + k0;
    FragAB u;
    u.q[0] = *(const uint4*)(p + 8 * half);
    u.q[1] = *(const uint4*)(p + 16 + 8 * half);
    return u.v;
}

// B fragment (32x16 bf16, KxN) loaded from B^T stored row-major [N][K]:
// lane r holds column n0+r, e<8 -> k = k0 + 8*half + e ; e>=8 -> k = k0 + 16 + 8*half + (e-8)
__device__ __forceinline__ v16bf load_bt_frag(const u16* __restrict__ Bt, int ldb,
                                              int n0, int k0, int lane) {
    int r = lane & 15, half = lane >> 4;
    const u16* p = Bt + (size_t)(n0 + r) * ldb + k0;
    FragAB u;
    u.q[0] = *(const uint4*)(p + 8 * half);
    u.q[1] = *(const uint4*)(p + 16 + 8 * half);
    return u.v;
}

// ---------------------------------------------------------------------------
// f32 [K][N] -> bf16 transposed [N][K]
__global__ void k_cvtT(const float* __restrict__ w, u16* __restrict__ wt,
                       int K, int N) {
    int total = K * N;
    for (int i = blockIdx.x * blockDim.x + threadIdx.x; i < total;
         i += gridDim.x * blockDim.x) {
        int n = i / K, k = i - n * K;
        wt[i] = f2bf(w[(size_t)k * N + n]);
    }
}

// x f32 -> xf f32 copy + xb bf16
__global__ void k_init(const float* __restrict__ x, float* __restrict__ xf,
                       u16* __restrict__ xb, int total) {
    for (int i = blockIdx.x * blockDim.x + threadIdx.x; i < total;
         i += gridDim.x * blockDim.x) {
        float v = x[i];
        xf[i] = v;
        xb[i] = f2bf(v);
    }
}

// ---------------------------------------------------------------------------
// Generic WMMA GEMM:  C = A(bf16 MxK row-major) * B(bf16 KxN, passed as B^T NxK)
// One wave -> one 32x32 tile (2x2 wmma). grid = (ceil(N/256), M/32), block 256.
// mode 0: f32 out[M][N] (+bias)        mode 1: bf16 out[M][N], bias+ReLU
// mode 2: bf16 qk[h][m][d], n=d*8+h    mode 3: bf16 vT[h][d][m], n=d*8+h
__global__ void k_gemm(const u16* __restrict__ A, const u16* __restrict__ Bt,
                       int M, int N, int K, const float* __restrict__ bias,
                       int mode, float* __restrict__ outf,
                       u16* __restrict__ outh) {
    int wave = threadIdx.x >> 5, lane = threadIdx.x & 31;
    int r = lane & 15, half = lane >> 4;
    int n0 = (blockIdx.x * 8 + wave) * 32;
    int m0 = blockIdx.y * 32;
    if (n0 >= N || m0 >= M) return;

    FragC acc[2][2];
#pragma unroll
    for (int mt = 0; mt < 2; ++mt)
#pragma unroll
        for (int nt = 0; nt < 2; ++nt)
#pragma unroll
            for (int i = 0; i < 8; ++i) acc[mt][nt].f[i] = 0.f;

    for (int k0 = 0; k0 < K; k0 += 32) {
        v16bf a0 = load_a_frag(A, K, m0,      k0, lane);
        v16bf a1 = load_a_frag(A, K, m0 + 16, k0, lane);
        v16bf b0 = load_bt_frag(Bt, K, n0,      k0, lane);
        v16bf b1 = load_bt_frag(Bt, K, n0 + 16, k0, lane);
        acc[0][0].v = wmma_bf16(a0, b0, acc[0][0].v);
        acc[0][1].v = wmma_bf16(a0, b1, acc[0][1].v);
        acc[1][0].v = wmma_bf16(a1, b0, acc[1][0].v);
        acc[1][1].v = wmma_bf16(a1, b1, acc[1][1].v);
    }

#pragma unroll
    for (int mt = 0; mt < 2; ++mt) {
#pragma unroll
        for (int nt = 0; nt < 2; ++nt) {
            FragC& c = acc[mt][nt];
            int mm = m0 + mt * 16;
            int n  = n0 + nt * 16 + r;
            if (mode == 0) {
                float bv = bias ? bias[n] : 0.f;
#pragma unroll
                for (int i = 0; i < 8; ++i)
                    outf[(size_t)(mm + i + 8 * half) * N + n] = c.f[i] + bv;
            } else if (mode == 1) {
                float bv = bias ? bias[n] : 0.f;
#pragma unroll
                for (int i = 0; i < 8; ++i) {
                    float v = c.f[i] + bv;
                    v = v > 0.f ? v : 0.f;   // leaky_relu slope 0 == relu
                    outh[(size_t)(mm + i + 8 * half) * N + n] = f2bf(v);
                }
            } else if (mode == 2) {          // [h][tok][d]
                int h = n & 7, d = n >> 3;
#pragma unroll
                for (int i = 0; i < 8; ++i)
                    outh[((size_t)h * NTOK + (mm + i + 8 * half)) * DM + d] =
                        f2bf(c.f[i]);
            } else {                         // mode 3: [h][d][tok]
                int h = n & 7, d = n >> 3;
                union { uint4 q; u16 s[8]; } pk;
#pragma unroll
                for (int i = 0; i < 8; ++i) pk.s[i] = f2bf(c.f[i]);
                u16* dst = outh + ((size_t)h * DM + d) * NTOK + mm + 8 * half;
                *(uint4*)dst = pk.q;
            }
        }
    }
}

// ---------------------------------------------------------------------------
// Flash attention.  grid = (NTOK/128, HEADS), block = 256 (8 waves).
// Wave w handles 16 query rows; loops keys in steps of 32.
// q,k: bf16 [h][tok][128]; vT: bf16 [h][128][tok]; o: bf16 [tok][d*8+h].
__global__ void k_attn(const u16* __restrict__ qb, const u16* __restrict__ kb,
                       const u16* __restrict__ vT, u16* __restrict__ o) {
    int h    = blockIdx.y;
    int wave = threadIdx.x >> 5, lane = threadIdx.x & 31;
    int r = lane & 15, half = lane >> 4;
    int q0 = blockIdx.x * 128 + wave * 16;

    const u16* Q  = qb + (size_t)h * NTOK * DM;
    const u16* Kh = kb + (size_t)h * NTOK * DM;
    const u16* Vh = vT + (size_t)h * DM * NTOK;

    // wave-private LDS tile for P: C-frag layout -> A-frag layout transpose
    __shared__ __align__(16) u16 pshare[8][16][40];   // 16 rows x 32 keys (+pad)

    v16bf qf[4];
#pragma unroll
    for (int dk = 0; dk < 4; ++dk) qf[dk] = load_a_frag(Q, DM, q0, dk * 32, lane);

    FragC of[8];
#pragma unroll
    for (int t = 0; t < 8; ++t)
#pragma unroll
        for (int i = 0; i < 8; ++i) of[t].f[i] = 0.f;

    float rmax[8], rsum[8];
#pragma unroll
    for (int i = 0; i < 8; ++i) { rmax[i] = -1e30f; rsum[i] = 0.f; }

    const float scale = 0.08838834764831845f;   // 1/sqrt(128)

    for (int kb0 = 0; kb0 < NTOK; kb0 += 32) {
        // ---- S = Q K^T for 32 keys (two 16-wide tiles) ----
        FragC s0, s1;
#pragma unroll
        for (int i = 0; i < 8; ++i) { s0.f[i] = 0.f; s1.f[i] = 0.f; }
#pragma unroll
        for (int dk = 0; dk < 4; ++dk) {
            v16bf b0 = load_bt_frag(Kh, DM, kb0,      dk * 32, lane);
            v16bf b1 = load_bt_frag(Kh, DM, kb0 + 16, dk * 32, lane);
            s0.v = wmma_bf16(qf[dk], b0, s0.v);
            s1.v = wmma_bf16(qf[dk], b1, s1.v);
        }

        // ---- online softmax (row i+8*half lives in f[i] across 16 lanes) ----
#pragma unroll
        for (int i = 0; i < 8; ++i) {
            float a0 = s0.f[i] * scale;
            float a1 = s1.f[i] * scale;
            float m = fmaxf(a0, a1);
#pragma unroll
            for (int off = 1; off < 16; off <<= 1)
                m = fmaxf(m, __shfl_xor(m, off, 32));
            float mn   = fmaxf(rmax[i], m);
            float corr = __expf(rmax[i] - mn);
            float p0 = __expf(a0 - mn);
            float p1 = __expf(a1 - mn);
            float ps = p0 + p1;
#pragma unroll
            for (int off = 1; off < 16; off <<= 1)
                ps += __shfl_xor(ps, off, 32);
            rsum[i] = rsum[i] * corr + ps;
            rmax[i] = mn;
#pragma unroll
            for (int t = 0; t < 8; ++t) of[t].f[i] *= corr;
            int row = i + 8 * half;
            pshare[wave][row][r]      = f2bf(p0);
            pshare[wave][row][16 + r] = f2bf(p1);
        }

        // ---- read P back as A fragment (DS ops in-order within a wave) ----
        FragAB pa;
        const u16* pp = &pshare[wave][r][0];
        pa.q[0] = *(const uint4*)(pp + 8 * half);
        pa.q[1] = *(const uint4*)(pp + 16 + 8 * half);

        // ---- O += P * V ----
#pragma unroll
        for (int dt = 0; dt < 8; ++dt) {
            v16bf vb = load_bt_frag(Vh, NTOK, dt * 16, kb0, lane);
            of[dt].v = wmma_bf16(pa.v, vb, of[dt].v);
        }
    }

    // ---- normalize + write interleaved o[tok][d*8+h] ----
#pragma unroll
    for (int i = 0; i < 8; ++i) {
        float inv = 1.f / rsum[i];
        int m = q0 + i + 8 * half;
#pragma unroll
        for (int dt = 0; dt < 8; ++dt) {
            int d = dt * 16 + r;
            o[(size_t)m * NQK + d * 8 + h] = f2bf(of[dt].f[i] * inv);
        }
    }
}

// ---------------------------------------------------------------------------
// y = LayerNorm(a + res) * w + b ; writes f32 and bf16 copies. grid=4096,block=128
__global__ void k_res_ln(const float* __restrict__ a, const float* __restrict__ res,
                         const float* __restrict__ w, const float* __restrict__ b,
                         float* __restrict__ y, u16* __restrict__ yb) {
    int m = blockIdx.x, t = threadIdx.x;
    __shared__ float sh[128];
    float v = a[(size_t)m * DM + t] + res[(size_t)m * DM + t];
    sh[t] = v; __syncthreads();
    for (int s = 64; s > 0; s >>= 1) { if (t < s) sh[t] += sh[t + s]; __syncthreads(); }
    float mu = sh[0] * (1.f / DM);
    __syncthreads();
    float d = v - mu;
    sh[t] = d * d; __syncthreads();
    for (int s = 64; s > 0; s >>= 1) { if (t < s) sh[t] += sh[t + s]; __syncthreads(); }
    float var = sh[0] * (1.f / DM);
    float o = d * rsqrtf(var + 1e-5f) * w[t] + b[t];
    y[(size_t)m * DM + t]  = o;
    yb[(size_t)m * DM + t] = f2bf(o);
}

__global__ void k_softplus(const float* __restrict__ x, float* __restrict__ y,
                           int total) {
    for (int i = blockIdx.x * blockDim.x + threadIdx.x; i < total;
         i += gridDim.x * blockDim.x) {
        float v = x[i];
        y[i] = (v > 20.f) ? v : log1pf(__expf(v));
    }
}

// ---------------------------------------------------------------------------
extern "C" void kernel_launch(void* const* d_in, const int* in_sizes, int n_in,
                              void* d_out, int out_size, void* d_ws, size_t ws_size,
                              hipStream_t stream) {
    (void)in_sizes; (void)n_in; (void)out_size; (void)ws_size;
    const float* x = (const float*)d_in[0];

    char* w = (char*)d_ws;
    size_t off = 0;
    auto alloc = [&](size_t bytes) -> void* {
        void* p = w + off;
        off = (off + bytes + 255) & ~(size_t)255;
        return p;
    };

    float* xf  = (float*)alloc((size_t)NTOK * DM * 4);
    float* tmp = (float*)alloc((size_t)NTOK * DM * 4);
    u16*   xb  = (u16*)  alloc((size_t)NTOK * DM * 2);
    u16*   hb  = (u16*)  alloc((size_t)NTOK * DM * 2);
    u16*   wT  = (u16*)  alloc((size_t)NQK * DM * 2);     // reused per weight
    u16*   qh  = (u16*)  alloc((size_t)HEADS * NTOK * DM * 2);
    u16*   kh  = (u16*)  alloc((size_t)HEADS * NTOK * DM * 2);
    u16*   vTb = (u16*)  alloc((size_t)HEADS * DM * NTOK * 2);
    u16*   ob  = (u16*)  alloc((size_t)NTOK * NQK * 2);

    k_init<<<512, 256, 0, stream>>>(x, xf, xb, NTOK * DM);

    for (int L = 0; L < 5; ++L) {
        const float* wq  = (const float*)d_in[1 + L * 12 + 0];
        const float* wk  = (const float*)d_in[1 + L * 12 + 1];
        const float* wv  = (const float*)d_in[1 + L * 12 + 2];
        const float* wo  = (const float*)d_in[1 + L * 12 + 3];
        const float* mlw = (const float*)d_in[1 + L * 12 + 4];
        const float* mlb = (const float*)d_in[1 + L * 12 + 5];
        const float* w1  = (const float*)d_in[1 + L * 12 + 6];
        const float* b1  = (const float*)d_in[1 + L * 12 + 7];
        const float* w2  = (const float*)d_in[1 + L * 12 + 8];
        const float* b2  = (const float*)d_in[1 + L * 12 + 9];
        const float* flw = (const float*)d_in[1 + L * 12 + 10];
        const float* flb = (const float*)d_in[1 + L * 12 + 11];

        // ---- MHA ----
        k_cvtT<<<512, 256, 0, stream>>>(wq, wT, DM, NQK);
        k_gemm<<<dim3(NQK / 256, NTOK / 32), 256, 0, stream>>>(
            xb, wT, NTOK, NQK, DM, nullptr, 2, nullptr, qh);
        k_cvtT<<<512, 256, 0, stream>>>(wk, wT, DM, NQK);
        k_gemm<<<dim3(NQK / 256, NTOK / 32), 256, 0, stream>>>(
            xb, wT, NTOK, NQK, DM, nullptr, 2, nullptr, kh);
        k_cvtT<<<512, 256, 0, stream>>>(wv, wT, DM, NQK);
        k_gemm<<<dim3(NQK / 256, NTOK / 32), 256, 0, stream>>>(
            xb, wT, NTOK, NQK, DM, nullptr, 3, nullptr, vTb);

        k_attn<<<dim3(NTOK / 128, HEADS), 256, 0, stream>>>(qh, kh, vTb, ob);

        k_cvtT<<<512, 256, 0, stream>>>(wo, wT, NQK, DM);
        k_gemm<<<dim3(1, NTOK / 32), 256, 0, stream>>>(
            ob, wT, NTOK, DM, NQK, nullptr, 0, tmp, nullptr);
        k_res_ln<<<NTOK, DM, 0, stream>>>(tmp, xf, mlw, mlb, xf, xb);

        // ---- FFN ----
        k_cvtT<<<512, 256, 0, stream>>>(w1, wT, DM, DM);
        k_gemm<<<dim3(1, NTOK / 32), 256, 0, stream>>>(
            xb, wT, NTOK, DM, DM, b1, 1, nullptr, hb);
        k_cvtT<<<512, 256, 0, stream>>>(w2, wT, DM, DM);
        k_gemm<<<dim3(1, NTOK / 32), 256, 0, stream>>>(
            hb, wT, NTOK, DM, DM, b2, 0, tmp, nullptr);
        k_res_ln<<<NTOK, DM, 0, stream>>>(tmp, xf, flw, flb, xf, xb);
    }

    k_softplus<<<512, 256, 0, stream>>>(xf, (float*)d_out, NTOK * DM);
}
